// LocalMultiHeadEncDecAttention_23742579212300
// MI455X (gfx1250) — matrix operand order repaired and checked by reading, not compiled, not run on hardware
//
#include <hip/hip_runtime.h>
#include <hip/hip_bf16.h>

// ---------- types for WMMA fragments ----------
typedef __attribute__((ext_vector_type(16))) __bf16 v16bf;
typedef __attribute__((ext_vector_type(8)))  float  v8f;

union F16x16 { v16bf v; unsigned int ui[8]; unsigned short us[16]; };
union F32x8  { v8f   v; float f[8]; };

__device__ __forceinline__ unsigned short f2bf(float f) {
    unsigned int u = __float_as_uint(f);
    u = (u + 0x7FFFu + ((u >> 16) & 1u)) >> 16;   // RNE
    return (unsigned short)u;
}

// =====================================================================
// GEMM: C[M x 512] = A[M x 512] @ W[512 x 512] + bias
//   A_BF16:   A is bf16 (ushort) else f32
//   OUT_PERM: write bf16 permuted [B,H=8,T,64] (n = h*64+k), else f32 row-major
// 64x64 tile / 256-thread block (8 waves); each wave does 16x32 via 2 WMMAs.
// =====================================================================
template<bool A_BF16, bool OUT_PERM>
__global__ __launch_bounds__(256) void gemm512_kernel(
    const void* __restrict__ Aptr, const float* __restrict__ W,
    const float* __restrict__ bias, void* __restrict__ Cptr,
    int M, int TQ)
{
    __shared__ unsigned short Atile[64][40];   // [m][k], pad 32->40 (conflict-free)
    __shared__ unsigned short Btile[64][40];   // [n][k], transposed W tile

    const int bm = blockIdx.x / 8;
    const int bn = blockIdx.x % 8;
    const int rowBase = bm * 64;
    const int colBase = bn * 64;

    const int tid  = threadIdx.x;
    const int lane = tid & 31;
    const int wv   = tid >> 5;          // wave 0..7
    const int wm   = (wv & 3) * 16;     // 16-row strip
    const int wn   = (wv >> 2) * 32;    // 32-col strip
    const int hi   = lane >> 4;
    const int mloc = lane & 15;

    const int ar = tid >> 2;            // 0..63  (A stage row)
    const int ac = (tid & 3) * 8;       // A stage col group
    const int kr = tid >> 3;            // 0..31  (W stage row)
    const int nc = (tid & 7) * 8;       // W stage col group

    F32x8 acc[2];
#pragma unroll
    for (int c = 0; c < 2; ++c)
#pragma unroll
        for (int r = 0; r < 8; ++r) acc[c].f[r] = 0.0f;

    for (int kk = 0; kk < 512; kk += 32) {
        // prefetch next tiles into GL2 while this one computes
        if (kk + 32 < 512) {
            if (A_BF16) {
                __builtin_prefetch(
                    (const unsigned short*)Aptr + (size_t)(rowBase + ar) * 512 + kk + 32 + ac, 0, 1);
            } else {
                __builtin_prefetch(
                    (const float*)Aptr + (size_t)(rowBase + ar) * 512 + kk + 32 + ac, 0, 1);
            }
            __builtin_prefetch(W + (size_t)(kk + 32 + kr) * 512 + colBase + nc, 0, 1);
        }

        // ---- stage A tile (64x32) as bf16 ----
        if (A_BF16) {
            const unsigned short* A = (const unsigned short*)Aptr;
            *(uint4*)&Atile[ar][ac] =
                *(const uint4*)(A + (size_t)(rowBase + ar) * 512 + kk + ac);
        } else {
            const float* A = (const float*)Aptr;
            const float* s = A + (size_t)(rowBase + ar) * 512 + kk + ac;
            float4 f0 = *(const float4*)s;
            float4 f1 = *(const float4*)(s + 4);
            uint4 pk;
            pk.x = (unsigned)f2bf(f0.x) | ((unsigned)f2bf(f0.y) << 16);
            pk.y = (unsigned)f2bf(f0.z) | ((unsigned)f2bf(f0.w) << 16);
            pk.z = (unsigned)f2bf(f1.x) | ((unsigned)f2bf(f1.y) << 16);
            pk.w = (unsigned)f2bf(f1.z) | ((unsigned)f2bf(f1.w) << 16);
            *(uint4*)&Atile[ar][ac] = pk;
        }
        // ---- stage W tile (32x64) transposed -> Btile[n][k] ----
        {
            const float* s = W + (size_t)(kk + kr) * 512 + colBase + nc;
            float4 f0 = *(const float4*)s;
            float4 f1 = *(const float4*)(s + 4);
            Btile[nc + 0][kr] = f2bf(f0.x);
            Btile[nc + 1][kr] = f2bf(f0.y);
            Btile[nc + 2][kr] = f2bf(f0.z);
            Btile[nc + 3][kr] = f2bf(f0.w);
            Btile[nc + 4][kr] = f2bf(f1.x);
            Btile[nc + 5][kr] = f2bf(f1.y);
            Btile[nc + 6][kr] = f2bf(f1.z);
            Btile[nc + 7][kr] = f2bf(f1.w);
        }
        __syncthreads();

        // ---- A fragment: 16x32, ISA layout ----
        F16x16 afrag;
        const int am = wm + mloc;
#pragma unroll
        for (int j = 0; j < 8; ++j) {
            const int k = 2 * (j & 3) + ((j >> 2) << 4) + (hi << 3);
            afrag.ui[j] = *(const unsigned int*)&Atile[am][k];
        }
#pragma unroll
        for (int ct = 0; ct < 2; ++ct) {
            F16x16 bfrag;
            const int n = wn + ct * 16 + mloc;
#pragma unroll
            for (int j = 0; j < 8; ++j) {
                const int k = 2 * (j & 3) + ((j >> 2) << 4) + (hi << 3);
                bfrag.ui[j] = *(const unsigned int*)&Btile[n][k];
            }
            acc[ct].v = __builtin_amdgcn_wmma_f32_16x16x32_bf16(
                false, afrag.v, false, bfrag.v, (short)0, acc[ct].v, false, false);
        }
        __syncthreads();
    }

    // ---- store C (bias fused) ----
#pragma unroll
    for (int ct = 0; ct < 2; ++ct) {
#pragma unroll
        for (int r = 0; r < 8; ++r) {
            const int m = rowBase + wm + r + hi * 8;
            const int n = colBase + wn + ct * 16 + mloc;
            const float val = acc[ct].f[r] + bias[n];
            if (OUT_PERM) {
                const int bb = m / TQ, t = m % TQ;
                const int hh = n >> 6, k = n & 63;
                ((unsigned short*)Cptr)[(((size_t)bb * 8 + hh) * TQ + t) * 64 + k] = f2bf(val);
            } else {
                ((float*)Cptr)[(size_t)m * 512 + n] = val;
            }
        }
    }
}

// =====================================================================
// Banded attention: one wave per (b, h, 16-query block).
// V tile is DMA'd into LDS with GLOBAL_LOAD_ASYNC_TO_LDS_B128 (ASYNCcnt),
// overlapping the score WMMAs; scores spill through LDS to keep VGPRs low;
// softmax normalization is folded in AFTER the ctx WMMAs (linear per row).
// =====================================================================
#define NKT 10
#define NKEY 160

__global__ __launch_bounds__(32) void attn_kernel(
    const unsigned short* __restrict__ qbf,
    const unsigned short* __restrict__ kbf,
    const unsigned short* __restrict__ vbf,
    unsigned short* __restrict__ ctx,
    const int* __restrict__ winPtr)
{
    __shared__ unsigned short Vlds[NKEY][64];   // [key][headdim]  (async DMA target)
    __shared__ float          Slds[16][160];    // masked, scaled scores
    __shared__ unsigned short Plds[16][168];    // unnormalized exp(P), padded stride

    const int TQ = 2048, TV = 2048, H = 8;
    const int qb = blockIdx.x % (TQ / 16);
    const int h  = (blockIdx.x / (TQ / 16)) % H;
    const int b  = blockIdx.x / ((TQ / 16) * H);

    const int lane = threadIdx.x;
    const int hi   = lane >> 4;
    const int mloc = lane & 15;

    const int window = *winPtr;
    const int half   = window >> 1;
    const float scale_c = (float)(TV - 1) / (float)(TQ - 1);
    const float invsq   = 0.125f;               // 1/sqrt(K=64)

    const int q0 = qb * 16;
    int ksBase = (int)rintf((float)q0 * scale_c) - half + 1;
    if (ksBase < 0) ksBase = 0;
    if (ksBase > TV - NKEY) ksBase = TV - NKEY;

    const size_t headOff = ((size_t)b * H + h) * TQ;

    // ---- async DMA of the 160x64 bf16 V tile straight into LDS ----
    // (generic LDS pointer's low 32 bits == LDS offset per aperture scheme)
#pragma unroll 4
    for (int it = 0; it < 40; ++it) {
        const int g   = it * 32 + lane;     // 0..1279 groups of 8 bf16
        const int row = g >> 3;
        const int cg  = (g & 7) * 8;
        unsigned ldsoff = (unsigned)(uintptr_t)&Vlds[row][cg];
        unsigned long long gaddr =
            (unsigned long long)(uintptr_t)(vbf + (headOff + ksBase + row) * 64 + cg);
        asm volatile("global_load_async_to_lds_b128 %0, %1, off"
                     :: "v"(ldsoff), "v"(gaddr) : "memory");
    }

    // ---- q fragments (2 head-dim chunks), direct aligned b32 loads ----
    F16x16 aq0, aq1;
    {
        const unsigned short* qrow = qbf + (headOff + q0 + mloc) * 64;
#pragma unroll
        for (int j = 0; j < 8; ++j) {
            const int k = 2 * (j & 3) + ((j >> 2) << 4) + (hi << 3);
            aq0.ui[j] = *(const unsigned int*)(qrow + k);
            aq1.ui[j] = *(const unsigned int*)(qrow + 32 + k);
        }
    }

    // ---- scores: 10 key tiles x (2 WMMA each); spill to LDS, track row max ----
    float mr[8];
#pragma unroll
    for (int r = 0; r < 8; ++r) mr[r] = -3.0e38f;

    for (int kt = 0; kt < NKT; ++kt) {
        F16x16 bk0, bk1;
        const int j = ksBase + kt * 16 + mloc;
        const unsigned short* krow = kbf + (headOff + j) * 64;
#pragma unroll
        for (int r = 0; r < 8; ++r) {
            const int kd = 2 * (r & 3) + ((r >> 2) << 4) + (hi << 3);
            bk0.ui[r] = *(const unsigned int*)(krow + kd);
            bk1.ui[r] = *(const unsigned int*)(krow + 32 + kd);
        }
        F32x8 acc;
#pragma unroll
        for (int r = 0; r < 8; ++r) acc.f[r] = 0.0f;
        acc.v = __builtin_amdgcn_wmma_f32_16x16x32_bf16(
            false, aq0.v, false, bk0.v, (short)0, acc.v, false, false);
        acc.v = __builtin_amdgcn_wmma_f32_16x16x32_bf16(
            false, aq1.v, false, bk1.v, (short)0, acc.v, false, false);
#pragma unroll
        for (int r = 0; r < 8; ++r) {
            const int   row = r + hi * 8;
            const int   qi  = q0 + row;
            const float center = rintf((float)qi * scale_c);
            float val = acc.f[r] * invsq;
            val = (fabsf((float)j - center) >= (float)half) ? -1e9f : val;
            Slds[row][kt * 16 + mloc] = val;
            mr[r] = fmaxf(mr[r], val);
        }
    }

    // ---- finish row max across the 16 columns of each half-wave ----
#pragma unroll
    for (int r = 0; r < 8; ++r)
#pragma unroll
        for (int off = 1; off < 16; off <<= 1)
            mr[r] = fmaxf(mr[r], __shfl_xor(mr[r], off, 32));

    // ---- exp pass: P = exp(s - max) (UNNORMALIZED, bf16); keep inv(sum) ----
    float inv[8];
#pragma unroll
    for (int r = 0; r < 8; ++r) {
        const int row = r + hi * 8;
        float sum = 0.0f;
        for (int kt = 0; kt < NKT; ++kt) {
            const float e = __expf(Slds[row][kt * 16 + mloc] - mr[r]);
            sum += e;
            Plds[row][kt * 16 + mloc] = f2bf(e);
        }
#pragma unroll
        for (int off = 1; off < 16; off <<= 1)
            sum += __shfl_xor(sum, off, 32);
        inv[r] = 1.0f / sum;
    }

    asm volatile("s_wait_asynccnt 0x0" ::: "memory");   // V tile landed in LDS
    __syncthreads();

    // ---- ctx = P(16x160) @ V(160x64): 4 col tiles x 5 key chunks ----
#pragma unroll
    for (int ct = 0; ct < 4; ++ct) {
        F32x8 acc;
#pragma unroll
        for (int r = 0; r < 8; ++r) acc.f[r] = 0.0f;
        for (int ch = 0; ch < 5; ++ch) {
            F16x16 ap, bvf;
            const unsigned short* prow = &Plds[mloc][0];
#pragma unroll
            for (int j = 0; j < 8; ++j) {
                const int k = ch * 32 + 2 * (j & 3) + ((j >> 2) << 4) + (hi << 3);
                ap.ui[j] = *(const unsigned int*)(prow + k);
            }
            const int hd = ct * 16 + mloc;
#pragma unroll
            for (int r = 0; r < 8; ++r) {
                const int j0 = ch * 32 + 2 * (r & 3) + ((r >> 2) << 4) + (hi << 3);
                const unsigned int lo  = Vlds[j0][hd];
                const unsigned int hi2 = Vlds[j0 + 1][hd];
                bvf.ui[r] = lo | (hi2 << 16);
            }
            acc.v = __builtin_amdgcn_wmma_f32_16x16x32_bf16(
                false, ap.v, false, bvf.v, (short)0, acc.v, false, false);
        }
#pragma unroll
        for (int r = 0; r < 8; ++r) {
            const int t   = q0 + r + hi * 8;
            const int col = h * 64 + ct * 16 + mloc;
            ctx[((size_t)b * TQ + t) * 512 + col] = f2bf(acc.f[r] * inv[r]);
        }
    }
}

// =====================================================================
extern "C" void kernel_launch(void* const* d_in, const int* in_sizes, int n_in,
                              void* d_out, int out_size, void* d_ws, size_t ws_size,
                              hipStream_t stream) {
    const float* query = (const float*)d_in[0];
    const float* value = (const float*)d_in[1];
    const float* Wq    = (const float*)d_in[2];
    const float* bq    = (const float*)d_in[3];
    const float* Wk    = (const float*)d_in[4];
    const float* bk    = (const float*)d_in[5];
    const float* Wv    = (const float*)d_in[6];
    const float* bv    = (const float*)d_in[7];
    const float* Wo    = (const float*)d_in[8];
    const float* bo    = (const float*)d_in[9];
    const int*   win   = (const int*)d_in[10];

    const int Bn = 2, TQ = 2048, H = 8, Kd = 64;
    const size_t projElems = (size_t)Bn * H * TQ * Kd;   // 2M bf16 each

    unsigned short* qbf = (unsigned short*)d_ws;
    unsigned short* kbf = qbf + projElems;
    unsigned short* vbf = kbf + projElems;
    unsigned short* ctx = vbf + projElems;               // B*TQ*512 bf16

    const int M = Bn * TQ;                               // 4096
    dim3 gblk(256), ggrid((M / 64) * (512 / 64));        // 512 blocks

    gemm512_kernel<false, true><<<ggrid, gblk, 0, stream>>>(query, Wq, bq, qbf, M, TQ);
    gemm512_kernel<false, true><<<ggrid, gblk, 0, stream>>>(value, Wk, bk, kbf, M, TQ);
    gemm512_kernel<false, true><<<ggrid, gblk, 0, stream>>>(value, Wv, bv, vbf, M, TQ);

    attn_kernel<<<dim3(Bn * H * (TQ / 16)), dim3(32), 0, stream>>>(qbf, kbf, vbf, ctx, win);

    gemm512_kernel<true, false><<<ggrid, gblk, 0, stream>>>(ctx, Wo, bo, d_out, M, TQ);
}
